// ISDALoss_13718125543890
// MI455X (gfx1250) — matrix operand order repaired and checked by reading, not compiled, not run on hardware
//
#include <hip/hip_runtime.h>
#include <hip/hip_bf16.h>

// ISDA loss, MI455X (gfx1250) version.
// Dominant cost: three fused N x C x A GEMMs (252 GFLOP) -> bf16 WMMA path.
// Register-blocked 2 N-tiles x 5 C-tiles per wave (C = 125 * 80 exactly) to
// lift arithmetic intensity vs L2 from ~10 FLOP/B to ~49 FLOP/B.
// W (160MB f32) stays L2-resident (192MB L2); converted f32->bf16 in-register.

#define N_ 1024
#define A_ 4096
#define C_ 10000

typedef __attribute__((ext_vector_type(16))) __bf16 v16bf;
typedef __attribute__((ext_vector_type(8)))  __bf16 v8bf;
typedef __attribute__((ext_vector_type(8)))  float  v8f;
typedef __attribute__((ext_vector_type(4)))  float  v4f;

__device__ __forceinline__ v16bf cat8(v8bf lo, v8bf hi) {
  v16bf r;
#pragma unroll
  for (int i = 0; i < 8; ++i) { r[i] = lo[i]; r[8 + i] = hi[i]; }
  return r;
}

// ---------------------------------------------------------------------------
// Kernel 1: per-sample CV row construction.
//   ave = sumf/denom, var = (sumf2 - sumf*ave)/denom  (count>=1 always)
//   w   = count/(count+Amount[k])
//   CVn = CoV*(1-w) + var*w + w*(1-w)*(Ave-ave)^2
// Emits bf16 CVn, CVn*Wk, features, and term3[n] = sum_a CVn*Wk^2.
// ---------------------------------------------------------------------------
__global__ __launch_bounds__(256) void isda_build_cv(
    const float* __restrict__ feat, const int* __restrict__ labels,
    const float* __restrict__ W, const float* __restrict__ AveM,
    const float* __restrict__ CoV, const float* __restrict__ Amount,
    __bf16* __restrict__ feat_bf, __bf16* __restrict__ cvn_bf,
    __bf16* __restrict__ cvnwk_bf, float* __restrict__ term3)
{
  __shared__ int   lab_sm[N_];
  __shared__ float red_sm[256];
  const int n   = blockIdx.x;
  const int tid = threadIdx.x;
  for (int i = tid; i < N_; i += 256) lab_sm[i] = labels[i];
  __syncthreads();

  const int k  = lab_sm[n];
  const int a0 = tid * 16;       // 256 threads * 16 = 4096 = A_

  float sumf[16], sumf2[16];
#pragma unroll
  for (int i = 0; i < 16; ++i) { sumf[i] = 0.f; sumf2[i] = 0.f; }

  int count = 0;
  for (int m = 0; m < N_; ++m) {
    if (lab_sm[m] == k) {          // uniform branch across the block
      ++count;
      const float* fr = feat + (long)m * A_ + a0;
#pragma unroll
      for (int j = 0; j < 4; ++j) {
        v4f v = *(const v4f*)(fr + 4 * j);
#pragma unroll
        for (int e = 0; e < 4; ++e) {
          float x = v[e];
          sumf[4 * j + e]  += x;
          sumf2[4 * j + e] += x * x;
        }
      }
    }
  }

  const float cnt     = (float)count;
  const float denom   = fmaxf(cnt, 1.f);
  const float inv_den = 1.f / denom;
  const float tot     = cnt + Amount[k];
  const float w       = (tot == 0.f) ? 0.f : (cnt / tot);
  const float omw     = 1.f - w;

  const float* Wr = W    + (long)k * A_ + a0;
  const float* Cr = CoV  + (long)k * A_ + a0;
  const float* Ar = AveM + (long)k * A_ + a0;
  const float* Fr = feat + (long)n * A_ + a0;

  float t3 = 0.f;
#pragma unroll
  for (int j = 0; j < 4; ++j) {
    v4f wv = *(const v4f*)(Wr + 4 * j);
    v4f cv = *(const v4f*)(Cr + 4 * j);
    v4f av = *(const v4f*)(Ar + 4 * j);
    v4f fv = *(const v4f*)(Fr + 4 * j);
#pragma unroll
    for (int e = 0; e < 4; ++e) {
      const int   i   = 4 * j + e;
      const float ave = sumf[i] * inv_den;
      const float var = (sumf2[i] - sumf[i] * ave) * inv_den;
      const float d   = av[e] - ave;
      const float cvn = cv[e] * omw + var * w + w * omw * d * d;
      const float wk  = wv[e];
      const long  off = (long)n * A_ + a0 + i;
      cvn_bf[off]   = (__bf16)cvn;
      cvnwk_bf[off] = (__bf16)(cvn * wk);
      feat_bf[off]  = (__bf16)fv[e];
      t3 += cvn * wk * wk;
    }
  }
  red_sm[tid] = t3;
  __syncthreads();
  for (int s = 128; s > 0; s >>= 1) {
    if (tid < s) red_sm[tid] += red_sm[tid + s];
    __syncthreads();
  }
  if (tid == 0) term3[n] = red_sm[0];
}

// ---------------------------------------------------------------------------
// Kernel 2: fused triple GEMM via v_wmma_f32_16x16x32_bf16.
// One wave computes a 32-row x 80-col block: 2 N-subtiles x 5 C-subtiles x
// 3 GEMMs = 30 accumulator chains.  Per K=32 step the 6 A operands are loaded
// once; each B subtile (f32 -> bf16 bw, bw2 in-register) feeds 6 WMMAs.
// A layout (16-bit 16x32): lane m=L&15, K = kh*16 + 0..15.
// B layout (32x16):        lane col=L&15, K = kh*16 + 0..15.
// Epilogue: y = accY + bias; aug = y + 0.5*ratio*(t1 - 2*t2 + term3[row]).
// ---------------------------------------------------------------------------
__global__ __launch_bounds__(32) void isda_gemm(
    const __bf16* __restrict__ feat_bf, const __bf16* __restrict__ cvn_bf,
    const __bf16* __restrict__ cvnwk_bf, const float* __restrict__ W,
    const float* __restrict__ bias, const float* __restrict__ term3,
    const int* __restrict__ ratio_p,
    float* __restrict__ y_out, float* __restrict__ aug)
{
  const int lane  = threadIdx.x;   // 0..31
  const int ctg   = blockIdx.x;    // 0..124  (80 columns each)
  const int ntg   = blockIdx.y;    // 0..31   (32 rows each)
  const int mloc  = lane & 15;
  const int khalf = lane >> 4;

  const __bf16* pa[2][3];
#pragma unroll
  for (int t = 0; t < 2; ++t) {
    const long arow = (long)(ntg * 32 + t * 16 + mloc) * A_ + khalf * 16;
    pa[t][0] = feat_bf  + arow;
    pa[t][1] = cvn_bf   + arow;
    pa[t][2] = cvnwk_bf + arow;
  }
  const float* pb[5];
#pragma unroll
  for (int c = 0; c < 5; ++c)
    pb[c] = W + (long)(ctg * 80 + c * 16 + mloc) * A_ + khalf * 16;

  v8f acc[2][5][3];
#pragma unroll
  for (int t = 0; t < 2; ++t)
#pragma unroll
    for (int c = 0; c < 5; ++c)
#pragma unroll
      for (int g = 0; g < 3; ++g) acc[t][c][g] = (v8f){};

  for (int kb = 0; kb < A_ / 32; ++kb) {
    const int kof = kb * 32;
    __builtin_prefetch(pb[0] + kof + 64, 0, 1);   // -> global_prefetch_b8

    v16bf a[2][3];
#pragma unroll
    for (int t = 0; t < 2; ++t)
#pragma unroll
      for (int g = 0; g < 3; ++g)
        a[t][g] = cat8(*(const v8bf*)(pa[t][g] + kof),
                       *(const v8bf*)(pa[t][g] + kof + 8));

#pragma unroll
    for (int c = 0; c < 5; ++c) {
      v16bf bw, bw2;
#pragma unroll
      for (int j = 0; j < 4; ++j) {
        v4f wv = *(const v4f*)(pb[c] + kof + 4 * j);
#pragma unroll
        for (int e = 0; e < 4; ++e) {
          bw [4 * j + e] = (__bf16)wv[e];
          bw2[4 * j + e] = (__bf16)(wv[e] * wv[e]);
        }
      }
#pragma unroll
      for (int t = 0; t < 2; ++t) {
        acc[t][c][0] = __builtin_amdgcn_wmma_f32_16x16x32_bf16(
            false, a[t][0], false, bw,  (short)0, acc[t][c][0], false, false);
        acc[t][c][1] = __builtin_amdgcn_wmma_f32_16x16x32_bf16(
            false, a[t][1], false, bw2, (short)0, acc[t][c][1], false, false);
        acc[t][c][2] = __builtin_amdgcn_wmma_f32_16x16x32_bf16(
            false, a[t][2], false, bw,  (short)0, acc[t][c][2], false, false);
      }
    }
  }

  const float ratio = (float)(*ratio_p);
#pragma unroll
  for (int c = 0; c < 5; ++c) {
    const int   colg = ctg * 80 + c * 16 + mloc;
    const float b    = bias[colg];
#pragma unroll
    for (int t = 0; t < 2; ++t) {
#pragma unroll
      for (int r = 0; r < 8; ++r) {
        const int   rowg = ntg * 32 + t * 16 + khalf * 8 + r; // C/D: M=kh*8+r
        const float yv   = acc[t][c][0][r] + b;
        const float s2   =
            ratio * (acc[t][c][1][r] - 2.f * acc[t][c][2][r] + term3[rowg]);
        const long  o    = (long)rowg * C_ + colg;
        y_out[o] = yv;
        aug[o]   = yv + 0.5f * s2;
      }
    }
  }
}

// ---------------------------------------------------------------------------
// Kernel 3: per-row log-softmax NLL.  loss_rows[n] = lse(aug[n,:]) - aug[n,y_n]
// ---------------------------------------------------------------------------
__global__ __launch_bounds__(256) void isda_softmax(
    const float* __restrict__ aug, const int* __restrict__ labels,
    float* __restrict__ loss_rows)
{
  __shared__ float sm[256];
  const int n = blockIdx.x, tid = threadIdx.x;
  const float* row = aug + (long)n * C_;

  float mx = -3.402823466e38f;
  for (int i = tid; i < C_; i += 256) mx = fmaxf(mx, row[i]);
  sm[tid] = mx; __syncthreads();
  for (int s = 128; s > 0; s >>= 1) {
    if (tid < s) sm[tid] = fmaxf(sm[tid], sm[tid + s]);
    __syncthreads();
  }
  mx = sm[0]; __syncthreads();

  float se = 0.f;
  for (int i = tid; i < C_; i += 256) se += __expf(row[i] - mx);
  sm[tid] = se; __syncthreads();
  for (int s = 128; s > 0; s >>= 1) {
    if (tid < s) sm[tid] += sm[tid + s];
    __syncthreads();
  }
  if (tid == 0) {
    const float lse = mx + __logf(sm[0]);
    loss_rows[n] = lse - row[labels[n]];
  }
}

__global__ __launch_bounds__(256) void isda_final(
    const float* __restrict__ loss_rows, float* __restrict__ out_loss)
{
  __shared__ float sm[256];
  const int tid = threadIdx.x;
  float s = 0.f;
  for (int i = tid; i < N_; i += 256) s += loss_rows[i];
  sm[tid] = s; __syncthreads();
  for (int k = 128; k > 0; k >>= 1) {
    if (tid < k) sm[tid] += sm[tid + k];
    __syncthreads();
  }
  if (tid == 0) out_loss[0] = sm[0] / (float)N_;
}

// ---------------------------------------------------------------------------
extern "C" void kernel_launch(void* const* d_in, const int* in_sizes, int n_in,
                              void* d_out, int out_size, void* d_ws, size_t ws_size,
                              hipStream_t stream) {
  (void)in_sizes; (void)n_in; (void)out_size; (void)ws_size;
  const float* features = (const float*)d_in[0];
  const int*   labels   = (const int*)  d_in[1];
  const float* fc_w     = (const float*)d_in[2];
  const float* fc_b     = (const float*)d_in[3];
  const float* AveM     = (const float*)d_in[4];
  const float* CoV      = (const float*)d_in[5];
  const float* Amount   = (const float*)d_in[6];
  const int*   ratio_p  = (const int*)  d_in[7];

  float* out    = (float*)d_out;
  float* loss_o = out;        // d_out[0]
  float* y_out  = out + 1;    // d_out[1..] = y [N,C]

  // Workspace layout (256B-aligned blocks).
  char* ws = (char*)d_ws;
  __bf16* feat_bf  = (__bf16*)(ws);                         // 8 MB
  __bf16* cvn_bf   = (__bf16*)(ws + 8388608);               // 8 MB
  __bf16* cvnwk_bf = (__bf16*)(ws + 16777216);              // 8 MB
  float*  term3    = (float*) (ws + 25165824);              // 4 KB
  float*  loss_r   = (float*) (ws + 25169920);              // 4 KB
  float*  aug      = (float*) (ws + 25174016);              // 40.96 MB

  isda_build_cv<<<N_, 256, 0, stream>>>(features, labels, fc_w, AveM, CoV,
                                        Amount, feat_bf, cvn_bf, cvnwk_bf,
                                        term3);
  isda_gemm<<<dim3(C_ / 80, N_ / 32), 32, 0, stream>>>(feat_bf, cvn_bf,
                                                       cvnwk_bf, fc_w, fc_b,
                                                       term3, ratio_p, y_out,
                                                       aug);
  isda_softmax<<<N_, 256, 0, stream>>>(aug, labels, loss_r);
  isda_final<<<1, 256, 0, stream>>>(loss_r, loss_o);
}
